// LstmSpeakerEncoder_23957327577376
// MI455X (gfx1250) — matrix operand order repaired and checked by reading, not compiled, not run on hardware
//
#include <hip/hip_runtime.h>
#include <hip/hip_bf16.h>

// ---------------------------------------------------------------------------
// 3-layer LSTM speaker encoder on gfx1250 (MI455X).
// Core math: bf16 WMMA 16x16x32 with f32 accumulation.
// All matrices pre-packed into WMMA-fragment-native layouts so GEMM loads are
// contiguous 32B-per-lane (global_load_b128 pairs).
// ---------------------------------------------------------------------------

typedef __attribute__((ext_vector_type(16))) __bf16 v16bf;
typedef __attribute__((ext_vector_type(8)))  float  v8f;

#define BB   256
#define TT   160
#define DIN  40
#define HH   768
#define G4H  3072
#define NKT_H 24          // 768 / 32 k-chunks for the recurrent part
#define TILE_HALF 512     // halves per 16x32 fragment tile (32 lanes * 16 halves)

// Fragment packing (CDNA5 ISA 7.12.2, 16-bit A 16x32 layout; B assumed
// symmetric over its K x 16 shape):
//   lanes 0-15  hold K = 0..7,16..23 ; lanes 16-31 hold K = 8..15,24..31
// rc = row index (A matrices) or column index (B matrix); k = K index.
__device__ __forceinline__ size_t frag_off(int rc, int k, int nkt) {
  int tmaj = rc >> 4;          // 16-wide tile index along rows/cols
  int mr   = rc & 15;
  int kt   = k >> 5;
  int kk   = k & 31;
  int grp  = (kk >> 3) & 1;
  int lane = grp * 16 + mr;
  int slot = (kk & 7) + ((kk & 16) ? 8 : 0);
  return ((size_t)tmaj * nkt + kt) * TILE_HALF + (size_t)lane * 16 + slot;
}

// ----------------------------- prep kernels --------------------------------

__global__ void zero_f32_kernel(float* p, int n) {
  int i = blockIdx.x * 256 + threadIdx.x;
  if (i < n) p[i] = 0.0f;
}

__global__ void zero_bf16_kernel(__hip_bfloat16* p, int n) {
  int i = blockIdx.x * 256 + threadIdx.x;
  if (i < n) p[i] = __float2bfloat16(0.0f);
}

// x [B,T,40] f32 -> per-t packed A matrices, K padded 40 -> 64 (nkt = 2)
__global__ void pack_x_kernel(const float* __restrict__ x,
                              __hip_bfloat16* __restrict__ xp) {
  int idx = blockIdx.x * 256 + threadIdx.x;      // B*T*64 total
  if (idx >= BB * TT * 64) return;
  int k  = idx & 63;
  int bt = idx >> 6;
  int t  = bt % TT;
  int b  = bt / TT;
  float v = (k < DIN) ? x[((size_t)b * TT + t) * DIN + k] : 0.0f;
  xp[(size_t)t * (16 * 2 * TILE_HALF) + frag_off(b, k, 2)] = __float2bfloat16(v);
}

// W = [Wi (kx rows, zero-padded to kxp) ; Wh (768 rows)], packed as WMMA-B.
__global__ void pack_w_kernel(const float* __restrict__ Wi,
                              const float* __restrict__ Wh,
                              __hip_bfloat16* __restrict__ wp,
                              int kx, int kxp, int nktW) {
  int total = (kxp + HH) * G4H;
  int idx = blockIdx.x * 256 + threadIdx.x;
  if (idx >= total) return;
  int col = idx % G4H;
  int r   = idx / G4H;
  float v;
  if (r < kxp) v = (r < kx) ? Wi[(size_t)r * G4H + col] : 0.0f;
  else         v = Wh[(size_t)(r - kxp) * G4H + col];
  wp[frag_off(col, r, nktW)] = __float2bfloat16(v);
}

// ------------------------------ step kernel --------------------------------
// One timestep: z = [x_t ; h_{t-1}] @ W + b, gates, cell/hidden update.
// Grid: 96 WGs = (256/64 batch blocks) x (768/32 hidden-col blocks).
// Block: 256 threads = 8 waves; wave w: gate = w>>1, row-half = w&1.
// Each wave: 2 row tiles x 2 col tiles -> 4 f32 accumulators.
template <int NKTX>
__global__ void lstm_step_kernel(const __hip_bfloat16* __restrict__ xp_t,   // packed A, nkt=NKTX
                                 const __hip_bfloat16* __restrict__ hp_prev,// packed A, nkt=24
                                 const __hip_bfloat16* __restrict__ wp,     // packed B, nkt=NKTX+24
                                 const float* __restrict__ bias,            // [4H]
                                 float* __restrict__ c,                     // [B*H]
                                 __hip_bfloat16* __restrict__ h_out)        // packed A, nkt=24
{
  constexpr int NKTW = NKTX + NKT_H;
  __shared__ float zs[64 * 128];               // 64 rows x (4 gates x 32 cols)

  const int wg   = blockIdx.x;
  const int mb   = wg & 3;                     // 64-row batch block
  const int nb   = wg >> 2;                    // 32-col hidden block (0..23)
  const int tid  = threadIdx.x;
  const int wave = tid >> 5;
  const int lane = tid & 31;
  const int gate = wave >> 1;
  const int half = wave & 1;

  const v8f zero8 = {0.f, 0.f, 0.f, 0.f, 0.f, 0.f, 0.f, 0.f};
  v8f acc[2][2];
  acc[0][0] = zero8; acc[0][1] = zero8; acc[1][0] = zero8; acc[1][1] = zero8;

  const v16bf* xpv = (const v16bf*)xp_t;
  const v16bf* hpv = (const v16bf*)hp_prev;
  const v16bf* wpv = (const v16bf*)wp;

  const int bt0 = mb * 4 + half * 2;           // first 16-row tile of this wave
  const int nt0 = gate * (HH / 16) + nb * 2;   // first 16-col tile (global over 4H)

  for (int kt = 0; kt < NKTW; ++kt) {
    v16bf a0, a1;
    if (kt < NKTX) {
      a0 = xpv[((size_t)(bt0 + 0) * NKTX + kt) * 32 + lane];
      a1 = xpv[((size_t)(bt0 + 1) * NKTX + kt) * 32 + lane];
    } else {
      int hk = kt - NKTX;
      a0 = hpv[((size_t)(bt0 + 0) * NKT_H + hk) * 32 + lane];
      a1 = hpv[((size_t)(bt0 + 1) * NKT_H + hk) * 32 + lane];
    }
    v16bf b0 = wpv[((size_t)(nt0 + 0) * NKTW + kt) * 32 + lane];
    v16bf b1 = wpv[((size_t)(nt0 + 1) * NKTW + kt) * 32 + lane];

    acc[0][0] = __builtin_amdgcn_wmma_f32_16x16x32_bf16(false, a0, false, b0,
                                                        (short)0, acc[0][0], false, false);
    acc[0][1] = __builtin_amdgcn_wmma_f32_16x16x32_bf16(false, a0, false, b1,
                                                        (short)0, acc[0][1], false, false);
    acc[1][0] = __builtin_amdgcn_wmma_f32_16x16x32_bf16(false, a1, false, b0,
                                                        (short)0, acc[1][0], false, false);
    acc[1][1] = __builtin_amdgcn_wmma_f32_16x16x32_bf16(false, a1, false, b1,
                                                        (short)0, acc[1][1], false, false);
  }

  // Scatter accumulators into the LDS gate buffer.
  // C/D layout: VGPR i holds M = i (lanes 0-15) or M = 8+i (lanes 16-31), N = lane%16.
  {
    const int rsel = (lane >> 4) << 3;         // +8 rows for lanes 16-31
    const int nlo  = lane & 15;
#pragma unroll
    for (int r = 0; r < 2; ++r) {
#pragma unroll
      for (int s = 0; s < 2; ++s) {
        int rbase = half * 32 + r * 16 + rsel;
        int col   = gate * 32 + s * 16 + nlo;
#pragma unroll
        for (int i = 0; i < 8; ++i)
          zs[(size_t)(rbase + i) * 128 + col] = acc[r][s][i];
      }
    }
  }
  __syncthreads();

  // Gate math: each thread owns 8 (row, col) cells of this WG's 64x32 region.
  const int row = tid >> 2;                    // 0..63
  const int gb  = (tid & 3) * 8;               // local col base
  const int brow = mb * 64 + row;              // global batch row
#pragma unroll
  for (int j = 0; j < 8; ++j) {
    int cl = gb + j;                           // 0..31 local hidden col
    int hc = nb * 32 + cl;                     // global hidden col
    float iv = zs[(size_t)row * 128 +  0 + cl] + bias[hc];
    float fv = zs[(size_t)row * 128 + 32 + cl] + bias[HH + hc];
    float gv = zs[(size_t)row * 128 + 64 + cl] + bias[2 * HH + hc];
    float ov = zs[(size_t)row * 128 + 96 + cl] + bias[3 * HH + hc];
    iv = 1.0f / (1.0f + __expf(-iv));
    fv = 1.0f / (1.0f + __expf(-fv));
    gv = 2.0f / (1.0f + __expf(-2.0f * gv)) - 1.0f;   // tanh
    ov = 1.0f / (1.0f + __expf(-ov));
    size_t ci = (size_t)brow * HH + hc;
    float cn = fv * c[ci] + iv * gv;
    c[ci] = cn;
    float tc = 2.0f / (1.0f + __expf(-2.0f * cn)) - 1.0f;
    h_out[frag_off(brow, hc, NKT_H)] = __float2bfloat16(ov * tc);
  }
}

// ------------------------------ mean kernel --------------------------------
__global__ void mean_kernel(const __hip_bfloat16* __restrict__ h2,
                            float* __restrict__ out) {
  int idx = blockIdx.x * 256 + threadIdx.x;    // B*H total
  if (idx >= BB * HH) return;
  int b  = idx / HH;
  int hc = idx % HH;
  size_t off = frag_off(b, hc, NKT_H);
  const size_t tstride = (size_t)16 * NKT_H * TILE_HALF;
  float s = 0.0f;
  for (int t = 0; t < TT; ++t)
    s += __bfloat162float(h2[(size_t)t * tstride + off]);
  out[idx] = s * (1.0f / TT);
}

// ------------------------------- host glue ---------------------------------
extern "C" void kernel_launch(void* const* d_in, const int* in_sizes, int n_in,
                              void* d_out, int out_size, void* d_ws, size_t ws_size,
                              hipStream_t stream) {
  (void)in_sizes; (void)n_in; (void)out_size; (void)ws_size;

  const float* x   = (const float*)d_in[0];
  const float* Wi[3] = {(const float*)d_in[1], (const float*)d_in[4], (const float*)d_in[7]};
  const float* Wh[3] = {(const float*)d_in[2], (const float*)d_in[5], (const float*)d_in[8]};
  const float* bi[3] = {(const float*)d_in[3], (const float*)d_in[6], (const float*)d_in[9]};
  float* out = (float*)d_out;

  // Workspace carve-up (256B aligned).
  uint8_t* ws = (uint8_t*)d_ws;
  size_t off = 0;
  auto carve = [&](size_t bytes) -> void* {
    void* p = ws + off;
    off = (off + bytes + 255) & ~(size_t)255;
    return p;
  };
  const int NT_TILES = G4H / 16;                       // 192 column tiles
  __hip_bfloat16* wp0 = (__hip_bfloat16*)carve((size_t)NT_TILES * 26 * TILE_HALF * 2);
  __hip_bfloat16* wp1 = (__hip_bfloat16*)carve((size_t)NT_TILES * 48 * TILE_HALF * 2);
  __hip_bfloat16* wp2 = (__hip_bfloat16*)carve((size_t)NT_TILES * 48 * TILE_HALF * 2);
  __hip_bfloat16* xp  = (__hip_bfloat16*)carve((size_t)TT * 16 * 2 * TILE_HALF * 2);
  __hip_bfloat16* hA  = (__hip_bfloat16*)carve((size_t)TT * 16 * NKT_H * TILE_HALF * 2);
  __hip_bfloat16* hB  = (__hip_bfloat16*)carve((size_t)TT * 16 * NKT_H * TILE_HALF * 2);
  __hip_bfloat16* zh  = (__hip_bfloat16*)carve((size_t)16 * NKT_H * TILE_HALF * 2);
  float*          cst = (float*)carve((size_t)BB * HH * 4);

  const size_t stepX0 = (size_t)16 * 2 * TILE_HALF;        // halves per t, layer-0 x
  const size_t stepH  = (size_t)16 * NKT_H * TILE_HALF;    // halves per t, packed h

  // Prep: zero h(-1), pack x and weights.
  zero_bf16_kernel<<<(BB * HH + 255) / 256, 256, 0, stream>>>(zh, BB * HH);
  pack_x_kernel<<<(BB * TT * 64 + 255) / 256, 256, 0, stream>>>(x, xp);
  pack_w_kernel<<<((64 + HH) * G4H + 255) / 256, 256, 0, stream>>>(Wi[0], Wh[0], wp0, DIN, 64, 26);
  pack_w_kernel<<<((HH + HH) * G4H + 255) / 256, 256, 0, stream>>>(Wi[1], Wh[1], wp1, HH, HH, 48);
  pack_w_kernel<<<((HH + HH) * G4H + 255) / 256, 256, 0, stream>>>(Wi[2], Wh[2], wp2, HH, HH, 48);

  const __hip_bfloat16* layer_in[3]  = {xp, hA, hB};
  __hip_bfloat16*       layer_out[3] = {hA, hB, hA};
  const __hip_bfloat16* wps[3]       = {wp0, wp1, wp2};

  for (int l = 0; l < 3; ++l) {
    zero_f32_kernel<<<(BB * HH + 255) / 256, 256, 0, stream>>>(cst, BB * HH);
    const size_t in_step = (l == 0) ? stepX0 : stepH;
    for (int t = 0; t < TT; ++t) {
      const __hip_bfloat16* xpt = layer_in[l] + (size_t)t * in_step;
      const __hip_bfloat16* hpv = (t == 0) ? zh : (layer_out[l] + (size_t)(t - 1) * stepH);
      __hip_bfloat16* hout = layer_out[l] + (size_t)t * stepH;
      if (l == 0)
        lstm_step_kernel<2><<<96, 256, 0, stream>>>(xpt, hpv, wps[l], bi[l], cst, hout);
      else
        lstm_step_kernel<24><<<96, 256, 0, stream>>>(xpt, hpv, wps[l], bi[l], cst, hout);
    }
  }

  mean_kernel<<<(BB * HH + 255) / 256, 256, 0, stream>>>(hA, out);
}
